// Memory_Block_42932493090858
// MI455X (gfx1250) — matrix operand order repaired and checked by reading, not compiled, not run on hardware
//
#include <hip/hip_runtime.h>

// ---------------------------------------------------------------------------
// VQ codebook EMA kernel for MI455X (gfx1250, wave32, WMMA).
// All GEMMs use v_wmma_f32_16x16x32_bf16 (bf16 inputs, f32 accumulate).
// Codebook tiles staged into LDS with global_load_async_to_lds_b128 (ASYNCcnt).
// ---------------------------------------------------------------------------

#define NTOK   65536
#define HDIM   512
#define KDIM   1024
#define RATE_F 0.999f

typedef __attribute__((ext_vector_type(16))) __bf16 bhalf16;
typedef __attribute__((ext_vector_type(8)))  __bf16 bhalf8;
typedef __attribute__((ext_vector_type(8)))  float  v8f;

__device__ __forceinline__ bhalf16 cat8(bhalf8 a, bhalf8 b) {
    bhalf16 r;
#pragma unroll
    for (int i = 0; i < 8; ++i) { r[i] = a[i]; r[i + 8] = b[i]; }
    return r;
}

// A-fragment (16x32 bf16): lane lo = row; element e -> K = e + (e<8?0:8) + hi*8
// -> two contiguous 16B chunks at k = hi*8 and k = 16 + hi*8.
__device__ __forceinline__ bhalf16 frag_a(const __bf16* p, int hi) {
    bhalf8 a0 = *(const bhalf8*)(p + hi * 8);
    bhalf8 a1 = *(const bhalf8*)(p + 16 + hi * 8);
    return cat8(a0, a1);
}

// B-fragment (32x16 bf16): lane lo = column N; element e -> K = e + (e<8?0:8) + hi*8
// -> two contiguous 16B chunks at k = hi*8 and k = 16 + hi*8.
__device__ __forceinline__ bhalf16 frag_b(const __bf16* p, int hi) {
    bhalf8 b0 = *(const bhalf8*)(p + hi * 8);
    bhalf8 b1 = *(const bhalf8*)(p + 16 + hi * 8);
    return cat8(b0, b1);
}

// Async DMA of 16 bytes per lane: global memory -> LDS, no VGPR round-trip.
// Tracked by ASYNCcnt. LDS dest byte-offset = low 32 bits of generic pointer
// (flat LDS aperture: addr[31:0] is the LDS address).
__device__ __forceinline__ void async_copy_b128(const void* gptr, void* lptr) {
    unsigned lds = (unsigned)(size_t)lptr;
    unsigned long long g = (unsigned long long)(size_t)gptr;
    asm volatile("global_load_async_to_lds_b128 %0, %1, off"
                 :: "v"(lds), "v"(g)
                 : "memory");
}
__device__ __forceinline__ void wait_async0() {
    asm volatile("s_wait_asynccnt 0" ::: "memory");
}

// ---------------------------------------------------------------------------
// K1: row-wise L2 normalize f32 -> bf16. One wave per row (8 rows / block).
// ---------------------------------------------------------------------------
__global__ __launch_bounds__(256) void normalize_rows_kernel(
    const float* __restrict__ src, __bf16* __restrict__ dst) {
    int wave = threadIdx.x >> 5, lane = threadIdx.x & 31;
    size_t row = (size_t)blockIdx.x * 8 + wave;
    const float* p = src + row * HDIM + lane * 16;
    float4 v0 = *(const float4*)(p + 0);
    float4 v1 = *(const float4*)(p + 4);
    float4 v2 = *(const float4*)(p + 8);
    float4 v3 = *(const float4*)(p + 12);
    float s = v0.x * v0.x + v0.y * v0.y + v0.z * v0.z + v0.w * v0.w +
              v1.x * v1.x + v1.y * v1.y + v1.z * v1.z + v1.w * v1.w +
              v2.x * v2.x + v2.y * v2.y + v2.z * v2.z + v2.w * v2.w +
              v3.x * v3.x + v3.y * v3.y + v3.z * v3.z + v3.w * v3.w;
#pragma unroll
    for (int off = 16; off > 0; off >>= 1) s += __shfl_xor(s, off, 32);
    float inv = 1.0f / fmaxf(sqrtf(s), 1e-12f);
    bhalf8 o0, o1;
    o0[0] = (__bf16)(v0.x * inv); o0[1] = (__bf16)(v0.y * inv);
    o0[2] = (__bf16)(v0.z * inv); o0[3] = (__bf16)(v0.w * inv);
    o0[4] = (__bf16)(v1.x * inv); o0[5] = (__bf16)(v1.y * inv);
    o0[6] = (__bf16)(v1.z * inv); o0[7] = (__bf16)(v1.w * inv);
    o1[0] = (__bf16)(v2.x * inv); o1[1] = (__bf16)(v2.y * inv);
    o1[2] = (__bf16)(v2.z * inv); o1[3] = (__bf16)(v2.w * inv);
    o1[4] = (__bf16)(v3.x * inv); o1[5] = (__bf16)(v3.y * inv);
    o1[6] = (__bf16)(v3.z * inv); o1[7] = (__bf16)(v3.w * inv);
    *(bhalf8*)(dst + row * HDIM + lane * 16)     = o0;
    *(bhalf8*)(dst + row * HDIM + lane * 16 + 8) = o1;
}

// ---------------------------------------------------------------------------
// K0: zero accumulators (counts + embed_sum are contiguous in ws).
// ---------------------------------------------------------------------------
__global__ __launch_bounds__(256) void zero_kernel(float* __restrict__ p, int n) {
    int i = blockIdx.x * 256 + threadIdx.x;
    if (i < n) p[i] = 0.0f;
}

// ---------------------------------------------------------------------------
// K2 / K5a: score GEMM  score = xn(64xC) @ mnB^T  tiled 64 rows x full KDIM.
// 8 waves = 4(M) x 2(N) 16x16 WMMA tiles, 32 codes per pass, K=C loop of 16
// chained v_wmma_f32_16x16x32_bf16. Argmax via packed u64 ds_max in LDS.
// Codebook tile staged via async-to-LDS DMA.
// Dynamic LDS: xs 64x520 bf16 | ms 32x520 bf16 | best u64[64]  (~100 KB).
// ---------------------------------------------------------------------------
template <bool DO_ARGMAX, bool WRITE_SCORE>
__global__ __launch_bounds__(256) void score_kernel(
    const __bf16* __restrict__ xn, const __bf16* __restrict__ mnB,
    int* __restrict__ embed_ind, float* __restrict__ score_out) {
    extern __shared__ __align__(16) char smem[];
    __bf16* xs = (__bf16*)smem;                 // 64 x 520
    __bf16* ms = xs + 64 * 520;                 // 32 x 520
    unsigned long long* best = (unsigned long long*)(ms + 32 * 520);

    int tid = threadIdx.x;
    int row0 = blockIdx.x * 64;

    // Stage x tile (64 x 512 bf16) once, via async DMA.
    for (int i = tid; i < 64 * 64; i += 256) {
        int r = i >> 6, c8 = i & 63;
        async_copy_b128(xn + (size_t)(row0 + r) * HDIM + c8 * 8,
                        &xs[r * 520 + c8 * 8]);
    }
    if (DO_ARGMAX && tid < 64) best[tid] = 0ULL;

    int wave = tid >> 5, lane = tid & 31;
    int lo = lane & 15, hi = lane >> 4;
    int mi = wave >> 1, ni = wave & 1;

    for (int np = 0; np < KDIM / 32; ++np) {
        __syncthreads();  // previous-pass readers done; also covers best init
        // Stage codebook tile (32 x 512 bf16) via async DMA.
        for (int i = tid; i < 32 * 64; i += 256) {
            int r = i >> 6, c8 = i & 63;
            async_copy_b128(mnB + (size_t)(np * 32 + r) * HDIM + c8 * 8,
                            &ms[r * 520 + c8 * 8]);
        }
        if (np + 1 < KDIM / 32)
            __builtin_prefetch(mnB + (size_t)(np + 1) * 32 * HDIM + tid * 64, 0, 1);
        wait_async0();
        __syncthreads();

        v8f acc = {};
        const __bf16* arow = xs + (mi * 16 + lo) * 520;
        const __bf16* brow = ms + (ni * 16 + lo) * 520;
#pragma unroll
        for (int kk = 0; kk < HDIM; kk += 32) {
            bhalf16 a = frag_a(arow + kk, hi);
            bhalf16 b = frag_b(brow + kk, hi);
            acc = __builtin_amdgcn_wmma_f32_16x16x32_bf16(
                false, a, false, b, (short)0, acc, false, false);
        }
        // C/D layout: VGPR r -> row = hi*8 + r, col = lo (within 16x16 tile).
        if (DO_ARGMAX) {
#pragma unroll
            for (int r = 0; r < 8; ++r) {
                unsigned u = __float_as_uint(acc[r]);
                u = (u & 0x80000000u) ? ~u : (u | 0x80000000u);  // order-preserving
                unsigned long long key =
                    ((unsigned long long)u << 32) |
                    (unsigned)(np * 32 + ni * 16 + lo);
                atomicMax(&best[mi * 16 + hi * 8 + r], key);
            }
        }
        if (WRITE_SCORE) {
#pragma unroll
            for (int r = 0; r < 8; ++r) {
                score_out[(size_t)(row0 + mi * 16 + hi * 8 + r) * KDIM +
                          np * 32 + ni * 16 + lo] = acc[r];
            }
        }
    }
    __syncthreads();
    if (DO_ARGMAX && tid < 64)
        embed_ind[row0 + tid] = (int)(best[tid] & 0xFFFFFFFFu);
}

// ---------------------------------------------------------------------------
// K3: scatter-add counts / embed_sum. 2 rows per block, 128 threads per row.
// ---------------------------------------------------------------------------
__global__ __launch_bounds__(256) void scatter_kernel(
    const float* __restrict__ x, const int* __restrict__ ind,
    const int* __restrict__ flag, float* __restrict__ counts,
    float* __restrict__ esum) {
    if (*flag == 0) return;
    int tid = threadIdx.x;
    int r = blockIdx.x * 2 + (tid >> 7);
    int c = (tid & 127) * 4;
    int k = ind[r];
    float4 v = *(const float4*)(x + (size_t)r * HDIM + c);
    float* dst = esum + (size_t)k * HDIM + c;
    atomicAdd(dst + 0, v.x);
    atomicAdd(dst + 1, v.y);
    atomicAdd(dst + 2, v.z);
    atomicAdd(dst + 3, v.w);
    if ((tid & 127) == 0) atomicAdd(counts + k, 1.0f);
}

// ---------------------------------------------------------------------------
// K4: EMA update + renormalize. One block per code row.
// update_flag==0 -> rate=1, scale=0 (m unchanged).
// ---------------------------------------------------------------------------
__global__ __launch_bounds__(256) void update_m_kernel(
    const float* __restrict__ m, const float* __restrict__ counts,
    const float* __restrict__ esum, const int* __restrict__ flag,
    __bf16* __restrict__ mnew_bf, __bf16* __restrict__ mn2_bf) {
    int k = blockIdx.x, tid = threadIdx.x;
    int upd = *flag;
    float rate = upd ? RATE_F : 1.0f;
    float cnt = counts[k];
    float scale = upd ? (1.0f - RATE_F) / (cnt + 1e-6f) : 0.0f;
    float2 mv = *(const float2*)(m + (size_t)k * HDIM + tid * 2);
    float2 sv = *(const float2*)(esum + (size_t)k * HDIM + tid * 2);
    float a = mv.x * rate + sv.x * scale;
    float b = mv.y * rate + sv.y * scale;
    float s = a * a + b * b;
#pragma unroll
    for (int off = 16; off > 0; off >>= 1) s += __shfl_xor(s, off, 32);
    __shared__ float red[8];
    __shared__ float invs;
    int wave = tid >> 5, lane = tid & 31;
    if (lane == 0) red[wave] = s;
    __syncthreads();
    if (tid == 0) {
        float t = 0.0f;
#pragma unroll
        for (int i = 0; i < 8; ++i) t += red[i];
        invs = 1.0f / fmaxf(sqrtf(t), 1e-12f);
    }
    __syncthreads();
    float inv = invs;
    size_t o = (size_t)k * HDIM + tid * 2;
    mnew_bf[o]     = (__bf16)a;
    mnew_bf[o + 1] = (__bf16)b;
    mn2_bf[o]      = (__bf16)(a * inv);
    mn2_bf[o + 1]  = (__bf16)(b * inv);
}

// ---------------------------------------------------------------------------
// K5b: per-row softmax stats (max, sum of exp). One wave per row.
// ---------------------------------------------------------------------------
__global__ __launch_bounds__(256) void row_stats_kernel(
    const float* __restrict__ score, float* __restrict__ rmax,
    float* __restrict__ rsum) {
    int wave = threadIdx.x >> 5, lane = threadIdx.x & 31;
    size_t row = (size_t)blockIdx.x * 8 + wave;
    const float* p = score + row * KDIM + (size_t)lane * 32;
    float v[32];
#pragma unroll
    for (int i = 0; i < 8; ++i) {
        float4 t = *(const float4*)(p + i * 4);
        v[i * 4 + 0] = t.x; v[i * 4 + 1] = t.y;
        v[i * 4 + 2] = t.z; v[i * 4 + 3] = t.w;
    }
    float mx = -3.0e38f;
#pragma unroll
    for (int i = 0; i < 32; ++i) mx = fmaxf(mx, v[i]);
#pragma unroll
    for (int off = 16; off > 0; off >>= 1) mx = fmaxf(mx, __shfl_xor(mx, off, 32));
    float sm = 0.0f;
#pragma unroll
    for (int i = 0; i < 32; ++i) sm += __expf(v[i] - mx);
#pragma unroll
    for (int off = 16; off > 0; off >>= 1) sm += __shfl_xor(sm, off, 32);
    if (lane == 0) { rmax[row] = mx; rsum[row] = sm; }
}

// ---------------------------------------------------------------------------
// K5c: out = softmax(score) @ m_new.  Block tile 64 rows x 256 cols, K=KDIM.
// A = exp(score - rowmax) converted to bf16 on stage, final scale by 1/rowsum.
// B staged transposed so fragments are contiguous ds_load_b128 pairs.
// ---------------------------------------------------------------------------
__global__ __launch_bounds__(256) void out_gemm_kernel(
    const float* __restrict__ score, const float* __restrict__ rmax,
    const float* __restrict__ rsum, const __bf16* __restrict__ mnew,
    float* __restrict__ out) {
    __shared__ __bf16 As[64 * 40];    // 64 rows x 32 k (padded)
    __shared__ __bf16 Bt[256 * 40];   // [n][k] transposed, 256 x 32 (padded)
    __shared__ float rmx[64];
    __shared__ float rsm[64];

    int tid = threadIdx.x;
    int row0 = blockIdx.x * 64;
    int col0 = blockIdx.y * 256;
    if (tid < 64) {
        rmx[tid] = rmax[row0 + tid];
        rsm[tid] = rsum[row0 + tid];
    }
    __syncthreads();

    int wave = tid >> 5, lane = tid & 31;
    int lo = lane & 15, hi = lane >> 4;
    int mi = wave >> 1, ni = wave & 1;   // rows mi*16, cols ni*16 + j*32

    v8f acc[8] = {};
    for (int kk = 0; kk < KDIM; kk += 32) {
        // Stage A: 64x32 f32 -> exp -> bf16. 8 elements / thread.
        {
            int r = tid >> 2;
            int c = (tid & 3) * 8;
            const float* sp = score + (size_t)(row0 + r) * KDIM + kk + c;
            float4 s0 = *(const float4*)(sp);
            float4 s1 = *(const float4*)(sp + 4);
            float mx = rmx[r];
            bhalf8 e;
            e[0] = (__bf16)__expf(s0.x - mx); e[1] = (__bf16)__expf(s0.y - mx);
            e[2] = (__bf16)__expf(s0.z - mx); e[3] = (__bf16)__expf(s0.w - mx);
            e[4] = (__bf16)__expf(s1.x - mx); e[5] = (__bf16)__expf(s1.y - mx);
            e[6] = (__bf16)__expf(s1.z - mx); e[7] = (__bf16)__expf(s1.w - mx);
            *(bhalf8*)(&As[r * 40 + c]) = e;
        }
        // Stage B transposed: m_new rows [kk,kk+32) x cols [col0,col0+256).
#pragma unroll
        for (int q = 0; q < 4; ++q) {
            int uid = tid * 4 + q;       // 0..1023 uint4s
            int r = uid >> 5;            // k-row 0..31
            int c8 = uid & 31;           // 8-col group
            uint4 v = *(const uint4*)(mnew + (size_t)(kk + r) * HDIM + col0 + c8 * 8);
            const __bf16* pv = (const __bf16*)&v;
#pragma unroll
            for (int i = 0; i < 8; ++i) Bt[(c8 * 8 + i) * 40 + r] = pv[i];
        }
        __syncthreads();

        bhalf16 a = frag_a(As + (mi * 16 + lo) * 40, hi);
#pragma unroll
        for (int j = 0; j < 8; ++j) {
            bhalf16 b = frag_b(Bt + (ni * 16 + j * 32 + lo) * 40, hi);
            acc[j] = __builtin_amdgcn_wmma_f32_16x16x32_bf16(
                false, a, false, b, (short)0, acc[j], false, false);
        }
        __syncthreads();
    }

#pragma unroll
    for (int r = 0; r < 8; ++r) {
        int lrow = mi * 16 + hi * 8 + r;
        float inv = __builtin_amdgcn_rcpf(rsm[lrow]);   // v_rcp_f32
#pragma unroll
        for (int j = 0; j < 8; ++j) {
            int col = col0 + ni * 16 + j * 32 + lo;
            out[(size_t)(row0 + lrow) * HDIM + col] = acc[j][r] * inv;
        }
    }
}

// ---------------------------------------------------------------------------
// Host-side launch.
// ---------------------------------------------------------------------------
extern "C" void kernel_launch(void* const* d_in, const int* in_sizes, int n_in,
                              void* d_out, int out_size, void* d_ws, size_t ws_size,
                              hipStream_t stream) {
    (void)in_sizes; (void)n_in; (void)out_size; (void)ws_size;
    const float* x    = (const float*)d_in[0];   // (N, C)
    const float* m    = (const float*)d_in[1];   // (K, C)
    const int*   flag = (const int*)d_in[2];     // scalar

    float* out_p   = (float*)d_out;                       // (N, C)
    float* score_p = out_p + (size_t)NTOK * HDIM;         // (N, K)

    // Workspace carve-up.
    char* ws = (char*)d_ws;
    __bf16* xn      = (__bf16*)ws;  ws += (size_t)NTOK * HDIM * 2;   // 64 MB
    __bf16* mn      = (__bf16*)ws;  ws += (size_t)KDIM * HDIM * 2;   // 1 MB
    __bf16* mn2     = (__bf16*)ws;  ws += (size_t)KDIM * HDIM * 2;   // 1 MB
    __bf16* mnew    = (__bf16*)ws;  ws += (size_t)KDIM * HDIM * 2;   // 1 MB
    int*    eind    = (int*)ws;     ws += (size_t)NTOK * 4;          // 256 KB
    float*  counts  = (float*)ws;   ws += (size_t)KDIM * 4;          // 4 KB
    float*  esum    = (float*)ws;   ws += (size_t)KDIM * HDIM * 4;   // 2 MB (adjacent to counts)
    float*  rmaxv   = (float*)ws;   ws += (size_t)NTOK * 4;
    float*  rsumv   = (float*)ws;   ws += (size_t)NTOK * 4;

    const size_t SMEM_SCORE = (size_t)(64 * 520 + 32 * 520) * 2 + 64 * 8; // ~100 KB

    // 1) normalize x, m -> bf16
    normalize_rows_kernel<<<NTOK / 8, 256, 0, stream>>>(x, xn);
    normalize_rows_kernel<<<KDIM / 8, 256, 0, stream>>>(m, mn);

    // 0) zero counts + embed_sum (contiguous)
    {
        int nz = KDIM + KDIM * HDIM;
        zero_kernel<<<(nz + 255) / 256, 256, 0, stream>>>(counts, nz);
    }

    // 2) score1 + argmax
    score_kernel<true, false><<<NTOK / 64, 256, SMEM_SCORE, stream>>>(
        xn, mn, eind, nullptr);

    // 3) scatter-add
    scatter_kernel<<<NTOK / 2, 256, 0, stream>>>(x, eind, flag, counts, esum);

    // 4) EMA update + renormalize
    update_m_kernel<<<KDIM, 256, 0, stream>>>(m, counts, esum, flag, mnew, mn2);

    // 5a) score2 -> output
    score_kernel<false, true><<<NTOK / 64, 256, SMEM_SCORE, stream>>>(
        xn, mn2, nullptr, score_p);

    // 5b) softmax stats
    row_stats_kernel<<<NTOK / 8, 256, 0, stream>>>(score_p, rmaxv, rsumv);

    // 5c) out = softmax(score2) @ m_new
    {
        dim3 grid(NTOK / 64, HDIM / 256);
        out_gemm_kernel<<<grid, 256, 0, stream>>>(score_p, rmaxv, rsumv, mnew, out_p);
    }
}
